// MambaBlock_1159641170368
// MI455X (gfx1250) — compile-verified, matrix operand order checked by reading, not run on hardware
//
#include <hip/hip_runtime.h>

#define D_MODEL 1024
#define D_STATE 16
#define D_CONV  4
#define D_INNER 2048
#define DT_RANK 64
#define DBC_W   (DT_RANK + 2*D_STATE)   // 96
#define BATCH   4
#define SEQ     2048
#define NROWS   (BATCH*SEQ)             // 8192

typedef __attribute__((ext_vector_type(16))) __bf16 v16bf;
typedef __attribute__((ext_vector_type(8)))  __bf16 v8bf;
typedef __attribute__((ext_vector_type(8)))  float  v8f;
typedef unsigned int u32x4 __attribute__((ext_vector_type(4)));
typedef int          i32x8 __attribute__((ext_vector_type(8)));
typedef int          i32x4 __attribute__((ext_vector_type(4)));

__device__ __forceinline__ unsigned short bf16_bits(float f) {
  union { float f; unsigned u; } v; v.f = f;
  return (unsigned short)((v.u + 0x7FFFu + ((v.u >> 16) & 1u)) >> 16); // RNE
}

// ---------------------------------------------------------------------------
// f32 -> bf16 bulk convert (n multiple of 256)
// ---------------------------------------------------------------------------
__launch_bounds__(256)
__global__ void cvt_bf16_kernel(const float* __restrict__ src,
                                unsigned short* __restrict__ dst, int n) {
  const int i = blockIdx.x * 256 + threadIdx.x;
  if (i < n) dst[i] = bf16_bits(src[i]);
}

// ---------------------------------------------------------------------------
// LayerNorm: one wave per row of 1024, 8 waves per block; bf16 output
// ---------------------------------------------------------------------------
__launch_bounds__(256)
__global__ void ln_kernel(const float* __restrict__ x, const float* __restrict__ g,
                          const float* __restrict__ b, unsigned short* __restrict__ xn) {
  const int lane = threadIdx.x & 31;
  const int wave = threadIdx.x >> 5;
  const int row  = blockIdx.x * 8 + wave;
  const float* xr = x + (size_t)row * D_MODEL;
  float v[32];
  float s = 0.f, s2 = 0.f;
  #pragma unroll
  for (int j = 0; j < 32; ++j) {
    float t = xr[lane + 32*j];
    v[j] = t; s += t; s2 += t*t;
  }
  #pragma unroll
  for (int m = 16; m >= 1; m >>= 1) {
    s  += __shfl_xor(s,  m);
    s2 += __shfl_xor(s2, m);
  }
  const float mu  = s * (1.f / D_MODEL);
  const float var = s2 * (1.f / D_MODEL) - mu * mu;
  const float rs  = rsqrtf(var + 1e-5f);
  unsigned short* o = xn + (size_t)row * D_MODEL;
  #pragma unroll
  for (int j = 0; j < 32; ++j) {
    int c = lane + 32*j;
    o[c] = bf16_bits((v[j] - mu) * rs * g[c] + b[c]);
  }
}

// ---------------------------------------------------------------------------
// bf16-WMMA GEMM: C[M,N] = A[M,K]*B[K,N] (+res).  A,B bf16; C,res f32.
// Block tile 128x128x32, 8 waves, wave tile 32x64 -> 8 WMMAs per K-step.
// A tile staged by the Tensor Data Mover (wave 0 issues TENSOR_LOAD_TO_LDS,
// TENSORcnt-waited); B tile staged transposed [n][k] via coalesced u16 loads.
// ---------------------------------------------------------------------------
#define BM 128
#define BN 128
#define BK 32

__launch_bounds__(256)
__global__ void wmma_gemm_kernel(const unsigned short* __restrict__ A,
                                 const unsigned short* __restrict__ B,
                                 float* __restrict__ C, const float* __restrict__ res,
                                 int M, int N, int K, int lda, int ldb, int ldc) {
  __shared__ __align__(16) unsigned short As[BM * BK];   // [m][k] bf16 (TDM dest)
  __shared__ __align__(16) unsigned short Bs[BN * BK];   // [n][k] bf16 (transposed)

  const int tid  = threadIdx.x;
  const int lane = tid & 31;
  const int wave = tid >> 5;        // 0..7
  const int wm   = wave & 3;        // 4 waves along M, 32 rows each
  const int wn   = wave >> 2;       // 2 waves along N, 64 cols each
  const int half = lane >> 4;
  const int r16  = lane & 15;

  const int mbase = blockIdx.y * BM;
  const int nbase = blockIdx.x * BN;

  // B staging: thread -> column b_n, k-group b_kg (16 k values)
  const int b_n  = tid & 127;
  const int b_kg = tid >> 7;        // 0..1
  const int gcB  = min(nbase + b_n, N - 1);   // clamp -> branch-free loads

  // TDM descriptor invariants (built once; only global_addr changes per step)
  const unsigned lds_a = (unsigned)(size_t)(void*)As;     // LDS byte offset
  const unsigned long long abase =
      (unsigned long long)(size_t)(const void*)A +
      (((unsigned long long)(unsigned)mbase * (unsigned long long)(unsigned)lda) << 1);

  v8f acc[2][4];
  #pragma unroll
  for (int i = 0; i < 2; ++i)
    #pragma unroll
    for (int j = 0; j < 4; ++j)
      acc[i][j] = (v8f){};

  for (int k0 = 0; k0 < K; k0 += BK) {
    // ---- stage A (128 rows x 32 cols bf16) via Tensor Data Mover ----
    if (tid < 32) {
      const unsigned long long ga = abase + ((unsigned long long)(unsigned)k0 << 1);
      u32x4 g0;
      g0[0] = 1u;                                          // count=1, user mode
      g0[1] = lds_a;                                       // lds_addr
      g0[2] = (unsigned)(ga & 0xFFFFFFFFull);              // global_addr[31:0]
      g0[3] = (unsigned)((ga >> 32) & 0x01FFFFFFull)       // global_addr[56:32]
            | 0x80000000u;                                 // type=2 ("image")
      i32x8 g1;
      g1[0] = 0x00010000;            // wg_mask=0, data_size=1 (2B), no pad/iter
      g1[1] = (int)(32u << 16);      // tensor_dim0[15:0]=32  (bits 63:48)
      g1[2] = (int)(128u << 16);     // tensor_dim0 hi=0 | tensor_dim1[15:0]=128
      g1[3] = (int)(32u << 16);      // tensor_dim1 hi=0 | tile_dim0=32
      g1[4] = 128;                   // tile_dim1=128, tile_dim2=0
      g1[5] = lda;                   // tensor_dim0_stride[31:0] (elements)
      g1[6] = 0;                     // stride hi / dim1_stride lo
      g1[7] = 0;
      const i32x4 gz4 = {0, 0, 0, 0};            // groups 2/3 unused (2D tile)
      const i32x8 gz8 = {0, 0, 0, 0, 0, 0, 0, 0};
      __builtin_amdgcn_tensor_load_to_lds(g0, g1, gz4, gz4, gz8, 0);
    }

    // ---- stage B (32 x 128) transposed: 16 coalesced bf16 column loads ----
    {
      unsigned short t[16];
      #pragma unroll
      for (int i = 0; i < 16; ++i)
        t[i] = B[(size_t)(k0 + b_kg * 16 + i) * ldb + gcB];
      uint4 p0, p1;
      p0.x = (unsigned)t[0]  | ((unsigned)t[1]  << 16);
      p0.y = (unsigned)t[2]  | ((unsigned)t[3]  << 16);
      p0.z = (unsigned)t[4]  | ((unsigned)t[5]  << 16);
      p0.w = (unsigned)t[6]  | ((unsigned)t[7]  << 16);
      p1.x = (unsigned)t[8]  | ((unsigned)t[9]  << 16);
      p1.y = (unsigned)t[10] | ((unsigned)t[11] << 16);
      p1.z = (unsigned)t[12] | ((unsigned)t[13] << 16);
      p1.w = (unsigned)t[14] | ((unsigned)t[15] << 16);
      *(uint4*)&Bs[b_n * BK + b_kg * 16    ] = p0;
      *(uint4*)&Bs[b_n * BK + b_kg * 16 + 8] = p1;
    }

    __builtin_amdgcn_s_wait_tensorcnt(0);   // no-op on waves 1..7
    __syncthreads();

    // ---- fragments + 8 WMMAs ----
    union F { v16bf v; v8bf h[2]; };
    F a[2], b[4];
    #pragma unroll
    for (int i = 0; i < 2; ++i) {
      const int rowa = wm * 32 + i * 16 + r16;
      a[i].h[0] = *(const v8bf*)&As[rowa * BK + half * 8];
      a[i].h[1] = *(const v8bf*)&As[rowa * BK + 16 + half * 8];
    }
    #pragma unroll
    for (int j = 0; j < 4; ++j) {
      const int coln = wn * 64 + j * 16 + r16;
      b[j].h[0] = *(const v8bf*)&Bs[coln * BK + half * 8];
      b[j].h[1] = *(const v8bf*)&Bs[coln * BK + 16 + half * 8];
    }
    #pragma unroll
    for (int i = 0; i < 2; ++i)
      #pragma unroll
      for (int j = 0; j < 4; ++j)
        acc[i][j] = __builtin_amdgcn_wmma_f32_16x16x32_bf16(
            false, a[i].v, false, b[j].v, (short)0, acc[i][j], false, false);
    __syncthreads();
  }

  // ---- epilogue: D layout lanes0-15: M=v,N=lane; lanes16-31: M=v+8 ----
  #pragma unroll
  for (int i = 0; i < 2; ++i) {
    const int rbase = mbase + wm * 32 + i * 16 + half * 8;
    #pragma unroll
    for (int j = 0; j < 4; ++j) {
      const int c = nbase + wn * 64 + j * 16 + r16;
      if (c < N) {
        #pragma unroll
        for (int vi = 0; vi < 8; ++vi) {
          const size_t idx = (size_t)(rbase + vi) * ldc + c;
          float o = acc[i][j][vi];
          if (res) o += res[idx];
          C[idx] = o;
        }
      }
    }
  }
}

// ---------------------------------------------------------------------------
// Causal depthwise conv (K=4) + SiLU; writes f32 (for scan) and bf16 (GEMM A)
// ---------------------------------------------------------------------------
__launch_bounds__(256)
__global__ void conv_silu_kernel(const float* __restrict__ xz, const float* __restrict__ cw,
                                 const float* __restrict__ cb, float* __restrict__ xc,
                                 unsigned short* __restrict__ xcb) {
  const int tid = blockIdx.x * 256 + threadIdx.x;   // (b,l,d) packed
  const int d = tid & (D_INNER-1);
  const int l = (tid >> 11) & (SEQ-1);
  const int b = tid >> 22;
  float acc = cb[d];
  #pragma unroll
  for (int k = 0; k < D_CONV; ++k) {
    int ll = l - (D_CONV-1) + k;
    if (ll >= 0)
      acc += xz[(size_t)(b*SEQ + ll) * (2*D_INNER) + d] * cw[d*D_CONV + k];
  }
  const float s = acc / (1.f + __expf(-acc));       // silu
  xc[(size_t)tid]  = s;
  xcb[(size_t)tid] = bf16_bits(s);
}

// ---------------------------------------------------------------------------
// dt = softplus(g3 + b_dt)
// ---------------------------------------------------------------------------
__launch_bounds__(256)
__global__ void softplus_kernel(const float* __restrict__ g3, const float* __restrict__ bdt,
                                float* __restrict__ dt) {
  const int tid = blockIdx.x * 256 + threadIdx.x;
  float v = g3[tid] + bdt[tid & (D_INNER-1)];
  dt[tid] = (v > 20.f) ? v : log1pf(__expf(v));
}

// ---------------------------------------------------------------------------
// Selective scan: one lane per (b,d,n); 16-lane shfl-xor reduce over D_STATE.
// Fuses +D*u and *silu(z); emits bf16 y (consumed only by GEMM4).
// ---------------------------------------------------------------------------
__launch_bounds__(256)
__global__ void scan_kernel(const float* __restrict__ dt, const float* __restrict__ dbc,
                            const float* __restrict__ u,  const float* __restrict__ xz,
                            const float* __restrict__ A_log, const float* __restrict__ Dp,
                            unsigned short* __restrict__ y, int Lsteps) {
  const int tid = blockIdx.x * 256 + threadIdx.x;
  const int n = tid & (D_STATE-1);
  const int d = (tid >> 4) & (D_INNER-1);
  const int b = tid >> 15;
  const float Aval = -__expf(A_log[d*D_STATE + n]);
  const float Dd   = Dp[d];
  float h = 0.f;
  #pragma clang loop unroll(disable)
  for (int t = 0; t < Lsteps; ++t) {
    const size_t row = (size_t)(b*SEQ + t);
    const float dtv = dt[row*D_INNER + d];
    const float uv  = u [row*D_INNER + d];
    const float Bv  = dbc[row*DBC_W + DT_RANK + n];
    const float Cv  = dbc[row*DBC_W + DT_RANK + D_STATE + n];
    const float dA  = __expf(dtv * Aval);
    h = dA * h + dtv * Bv * uv;
    float p = h * Cv;
    p += __shfl_xor(p, 1);
    p += __shfl_xor(p, 2);
    p += __shfl_xor(p, 4);
    p += __shfl_xor(p, 8);
    if (n == 0) {
      const float zv = xz[row*(2*D_INNER) + D_INNER + d];
      const float sz = zv / (1.f + __expf(-zv));
      y[row*D_INNER + d] = bf16_bits((p + Dd * uv) * sz);
    }
  }
}

// ---------------------------------------------------------------------------
extern "C" void kernel_launch(void* const* d_in, const int* in_sizes, int n_in,
                              void* d_out, int out_size, void* d_ws, size_t ws_size,
                              hipStream_t stream) {
  (void)in_sizes; (void)n_in; (void)out_size; (void)ws_size;
  const float* x      = (const float*)d_in[0];
  const float* ln_g   = (const float*)d_in[1];
  const float* ln_b   = (const float*)d_in[2];
  const float* W_in   = (const float*)d_in[3];
  const float* conv_w = (const float*)d_in[4];
  const float* conv_b = (const float*)d_in[5];
  const float* W_x    = (const float*)d_in[6];
  const float* W_dt   = (const float*)d_in[7];
  const float* b_dt   = (const float*)d_in[8];
  const float* A_log  = (const float*)d_in[9];
  const float* Dp     = (const float*)d_in[10];
  const float* W_out  = (const float*)d_in[11];
  float* out = (float*)d_out;

  // ---- workspace layout: f32 first, then bf16 ----
  float* ws   = (float*)d_ws;
  float* xz   = ws;                                    // 8192*4096
  float* xc   = xz  + (size_t)NROWS * 2*D_INNER;       // 8192*2048
  float* dbc  = xc  + (size_t)NROWS * D_INNER;         // 8192*96
  float* dtb  = dbc + (size_t)NROWS * DBC_W;           // 8192*2048
  unsigned short* h16  = (unsigned short*)(dtb + (size_t)NROWS * D_INNER);
  unsigned short* xnb  = h16;                          // 8192*1024
  unsigned short* xcb  = xnb  + (size_t)NROWS * D_MODEL;       // 8192*2048
  unsigned short* dbcb = xcb  + (size_t)NROWS * D_INNER;       // 8192*96
  unsigned short* ybb  = dbcb + (size_t)NROWS * DBC_W;         // 8192*2048
  unsigned short* winb = ybb  + (size_t)NROWS * D_INNER;       // 1024*4096
  unsigned short* wxb  = winb + (size_t)D_MODEL * 2*D_INNER;   // 2048*96
  unsigned short* wdtb = wxb  + (size_t)D_INNER * DBC_W;       // 64*2048
  unsigned short* woutb= wdtb + (size_t)DT_RANK * D_INNER;     // 2048*1024

  // 0) one-shot weight conversions f32 -> bf16
  cvt_bf16_kernel<<<(D_MODEL*2*D_INNER)/256, 256, 0, stream>>>(W_in,  winb, D_MODEL*2*D_INNER);
  cvt_bf16_kernel<<<(D_INNER*DBC_W)/256,     256, 0, stream>>>(W_x,   wxb,  D_INNER*DBC_W);
  cvt_bf16_kernel<<<(DT_RANK*D_INNER)/256,   256, 0, stream>>>(W_dt,  wdtb, DT_RANK*D_INNER);
  cvt_bf16_kernel<<<(D_INNER*D_MODEL)/256,   256, 0, stream>>>(W_out, woutb, D_INNER*D_MODEL);

  // 1) LayerNorm -> bf16
  ln_kernel<<<NROWS/8, 256, 0, stream>>>(x, ln_g, ln_b, xnb);

  // 2) xz = xn @ W_in   (8192x1024 x 1024x4096)
  wmma_gemm_kernel<<<dim3((2*D_INNER+BN-1)/BN, NROWS/BM), 256, 0, stream>>>(
      xnb, winb, xz, nullptr, NROWS, 2*D_INNER, D_MODEL, D_MODEL, 2*D_INNER, 2*D_INNER);

  // 3) xc = silu(causal_conv(xh)) -> f32 + bf16
  conv_silu_kernel<<<(NROWS*D_INNER)/256, 256, 0, stream>>>(xz, conv_w, conv_b, xc, xcb);

  // 4) dbc = xc @ W_x   (8192x2048 x 2048x96)
  wmma_gemm_kernel<<<dim3((DBC_W+BN-1)/BN, NROWS/BM), 256, 0, stream>>>(
      xcb, wxb, dbc, nullptr, NROWS, DBC_W, D_INNER, D_INNER, DBC_W, DBC_W);

  // 4b) dbc -> bf16 (GEMM3's A view: first 64 cols, lda=96)
  cvt_bf16_kernel<<<(NROWS*DBC_W)/256, 256, 0, stream>>>(dbc, dbcb, NROWS*DBC_W);

  // 5) dtb = dbc[:, :64] @ W_dt  (8192x64 x 64x2048)
  wmma_gemm_kernel<<<dim3((D_INNER+BN-1)/BN, NROWS/BM), 256, 0, stream>>>(
      dbcb, wdtb, dtb, nullptr, NROWS, D_INNER, DT_RANK, DBC_W, D_INNER, D_INNER);

  // 6) dt = softplus(dtb + b_dt)
  softplus_kernel<<<(NROWS*D_INNER)/256, 256, 0, stream>>>(dtb, b_dt, dtb);

  // 7) selective scan (fuses +D*u and *silu(z)) -> bf16 y
  scan_kernel<<<(BATCH*D_INNER*D_STATE)/256, 256, 0, stream>>>(
      dtb, dbc, xc, xz, A_log, Dp, ybb, SEQ);

  // 8) out = x + y @ W_out  (residual fused into epilogue)
  wmma_gemm_kernel<<<dim3((D_MODEL+BN-1)/BN, NROWS/BM), 256, 0, stream>>>(
      ybb, woutb, out, x, NROWS, D_MODEL, D_INNER, D_INNER, D_MODEL, D_MODEL);
}
